// MultiHeadAttention_16088947490887
// MI455X (gfx1250) — matrix operand order repaired and checked by reading, not compiled, not run on hardware
//
#include <hip/hip_runtime.h>
#include <hip/hip_bf16.h>

typedef __attribute__((ext_vector_type(16))) _Float16 v16h;
typedef __attribute__((ext_vector_type(8)))  _Float16 v8h;
typedef __attribute__((ext_vector_type(8)))  float    v8f;
typedef __attribute__((ext_vector_type(4)))  float    v4f;
typedef __attribute__((ext_vector_type(4)))  unsigned int u32x4;
typedef __attribute__((ext_vector_type(8)))  int      i32x8;
typedef __attribute__((ext_vector_type(4)))  int      i32x4;

namespace cfg {
constexpr int B = 2, S = 2048, D = 1024, H = 16, AD = 64;
constexpr int BS = B * S;        // 4096 merged rows
constexpr int KT = 64;           // attention key tile
constexpr int NW = 4;            // waves per workgroup (1 per SIMD32)
constexpr int LP = S + 8;        // padded logits row stride
}

#if defined(__HIP_DEVICE_COMPILE__) &&                        \
    __has_builtin(__builtin_amdgcn_tensor_load_to_lds) &&     \
    __has_builtin(__builtin_amdgcn_s_wait_tensorcnt)
#define USE_TDM 1
#else
#define USE_TDM 0
#endif

// D(16x16 f32) = A(16x32 f16) * B(32x16 f16) + C  — CDNA5 WMMA, wave32
__device__ __forceinline__ v8f wmma16(v16h a, v16h b, v8f c) {
  return __builtin_amdgcn_wmma_f32_16x16x32_f16(false, a, false, b, (short)0, c,
                                                false, false);
}

#if USE_TDM
// TDM 2-D f16 tile DMA: tile_d1 rows of tile_d0 halves, global row stride
// `stride0` (elements). LDS destination padded by TDM: `pad_amount` dwords
// after every `pad_interval` dwords (codes per D# group1 spec).
__device__ __forceinline__ void tdm_load_2d_f16(unsigned lds_byte_off,
                                                const _Float16* gsrc,
                                                int tile_d0, int tile_d1,
                                                int stride0, int pad_int_code,
                                                int pad_amt_code) {
  const unsigned long long ga = (unsigned long long)(size_t)gsrc;
  u32x4 g0;
  g0[0] = 1u;                                      // count=1 (valid user D#)
  g0[1] = lds_byte_off;                            // lds_addr
  g0[2] = (unsigned)ga;                            // global_addr[31:0]
  g0[3] = (unsigned)((ga >> 32) & 0x01ffffffu)     // global_addr[56:32]
          | (2u << 30);                            // type=2 ("image")
  i32x8 g1;
  g1[0] = (1 << 16)                                // data_size = 2 bytes
        | (1 << 20)                                // pad_enable
        | (pad_int_code << 22)                     // dwords between pads
        | (pad_amt_code << 25);                    // pad dwords
  g1[1] = (tile_d0 & 0xffff) << 16;                // tensor_dim0 lo16
  g1[2] = (tile_d1 & 0xffff) << 16;                // dim0 hi | tensor_dim1 lo16
  g1[3] = (tile_d0 & 0xffff) << 16;                // dim1 hi | tile_dim0
  g1[4] = (tile_d1 & 0xffff);                      // tile_dim1 | tile_dim2=0
  g1[5] = stride0;                                 // tensor_dim0_stride lo32
  g1[6] = 0;
  g1[7] = 0;
  const i32x4 z4 = {};
#if __clang_major__ >= 23
  const i32x8 z8 = {};
  __builtin_amdgcn_tensor_load_to_lds(g0, g1, z4, z4, z8, 0);
#else
  __builtin_amdgcn_tensor_load_to_lds(g0, g1, z4, z4, 0);
#endif
}
#endif  // USE_TDM

// ---------------------------------------------------------------------------
// QKV projection: head[b,h,:,:] (+= scatter strides) = (X @ W) * scale.
// 64x64 tile / workgroup, 4 waves, K-step 32. W tile stored TRANSPOSED in LDS
// so B-fragments are two contiguous 8-half (b128) reads per lane, at k = 8h and 16 + 8h.
// dStride/sStride select head layout: Q/K -> (S, AD); V -> (AD, S) transposed
// so the attention kernel can DMA V tiles with the TDM (no LDS transpose).
// ---------------------------------------------------------------------------
__global__ __launch_bounds__(128) void qkv_proj_kernel(
    const float* __restrict__ X, const float* __restrict__ W,
    _Float16* __restrict__ OutHead, float scale, int dStride, int sStride) {
  using namespace cfg;
  __shared__ _Float16 As[64][40];    // 64x32 A tile, row-major (padded)
  __shared__ _Float16 BsT[64][40];   // 32x64 W tile, TRANSPOSED (n, k)
  const int m0 = blockIdx.y * 64;
  const int n0 = blockIdx.x * 64;
  const int lane = threadIdx.x & 31;
  const int wave = threadIdx.x >> 5;
  const int half = lane >> 4;
  const int l16 = lane & 15;

  v8f acc[4] = {};

  for (int k0 = 0; k0 < D; k0 += 32) {
    __syncthreads();
    // Batch all global loads first (MLP), then convert/store to LDS.
    v4f xa[4], wa[4];
#pragma unroll
    for (int it = 0; it < 4; ++it) {
      const int i = threadIdx.x + it * 128;
      xa[it] = *(const v4f*)&X[(size_t)(m0 + (i >> 3)) * D + k0 + (i & 7) * 4];
      wa[it] =
          *(const v4f*)&W[(size_t)(k0 + (i >> 4)) * D + n0 + (i & 15) * 4];
    }
#pragma unroll
    for (int it = 0; it < 4; ++it) {
      const int i = threadIdx.x + it * 128;
      const int ar = i >> 3, ac = (i & 7) * 4;
      As[ar][ac + 0] = (_Float16)xa[it][0];
      As[ar][ac + 1] = (_Float16)xa[it][1];
      As[ar][ac + 2] = (_Float16)xa[it][2];
      As[ar][ac + 3] = (_Float16)xa[it][3];
      const int br = i >> 4, bc = (i & 15) * 4;  // k row, n col
      BsT[bc + 0][br] = (_Float16)wa[it][0];     // transpose store
      BsT[bc + 1][br] = (_Float16)wa[it][1];
      BsT[bc + 2][br] = (_Float16)wa[it][2];
      BsT[bc + 3][br] = (_Float16)wa[it][3];
    }
    __syncthreads();

    const int arow = wave * 16 + l16;
    const int koff = half * 8;
    v16h af;
#pragma unroll
    for (int e = 0; e < 8; ++e) {
      af[e]     = As[arow][koff + e];
      af[e + 8] = As[arow][16 + koff + e];
    }
#pragma unroll
    for (int nt = 0; nt < 4; ++nt) {
      const int bcol = nt * 16 + l16;
      const int kb = half * 8;
      v16h bf;
#pragma unroll
      for (int e = 0; e < 16; ++e) bf[e] = BsT[bcol][kb + (e & 7) + (e >> 3) * 16];  // table order
      acc[nt] = wmma16(af, bf, acc[nt]);
    }
  }

#pragma unroll
  for (int nt = 0; nt < 4; ++nt) {
#pragma unroll
    for (int r = 0; r < 8; ++r) {
      const int m = m0 + wave * 16 + half * 8 + r;
      const int n = n0 + nt * 16 + l16;
      const int b = m >> 11;          // m / S
      const int s = m & (S - 1);
      const int h = n >> 6;           // n / AD
      const int d = n & (AD - 1);
      OutHead[(size_t)(b * H + h) * (S * AD) + (size_t)d * dStride +
              (size_t)s * sStride] = (_Float16)(acc[nt][r] * scale);
    }
  }
}

// ---------------------------------------------------------------------------
// Attention: per (b,h), 64 q-rows / workgroup (16 per wave). Full logit rows
// live in LDS (f16): softmax needs no HBM round-trip; attn written once.
// K and V tiles staged by the Tensor Data Mover (D# pad == LDS row padding).
// V head tensor is pre-transposed (AD, S) so the V tile needs no transpose.
// ---------------------------------------------------------------------------
__global__ __launch_bounds__(cfg::NW * 32) void attention_kernel(
    const _Float16* __restrict__ Qh, const _Float16* __restrict__ Kh,
    const _Float16* __restrict__ VhT, float* __restrict__ Attn,
    _Float16* __restrict__ Ctx) {
  using namespace cfg;
  __shared__ _Float16 Lg[NW][16][LP];    // 4 x 16 x 2056 f16 = 257 KB
  __shared__ _Float16 Kt[KT][AD + 8];    // key tile (key, d), +16B/row pad
  __shared__ _Float16 Vt[AD][KT + 8];    // value tile (d, key), +16B/row pad
  __shared__ float RowMax[NW][16];
  __shared__ float RowSum[NW][16];

  const int bh = blockIdx.y;             // b*H + h
  const int q0 = blockIdx.x * (NW * 16);
  const int lane = threadIdx.x & 31;
  const int wave = threadIdx.x >> 5;
  const int half = lane >> 4;
  const int l16 = lane & 15;

  const size_t head = (size_t)bh * S * AD;
  const _Float16* Qp = Qh + head;
  const _Float16* Kp = Kh + head;
  const _Float16* Vp = VhT + head;       // (AD, S) layout

  // ---- Q tile (16x64) as two A fragments (contraction K=0..31 / 32..63)
  const int qrow = q0 + wave * 16 + l16;
  v16h aq0, aq1;
  {
    const v8h* qv = (const v8h*)(Qp + (size_t)qrow * AD);
    const v8h c0 = qv[half], c1 = qv[2 + half], c2 = qv[4 + half],
              c3 = qv[6 + half];
#pragma unroll
    for (int e = 0; e < 8; ++e) {
      aq0[e] = c0[e]; aq0[e + 8] = c1[e];
      aq1[e] = c2[e]; aq1[e + 8] = c3[e];
    }
  }

  // ---- Phase 1: logits = Qs Kᵀ (scale folded into Q projection)
  for (int kt = 0; kt < S / KT; ++kt) {
    __syncthreads();
#if USE_TDM
    if (wave == 0) {
      tdm_load_2d_f16((unsigned)(size_t)&Kt[0][0], Kp + (size_t)kt * KT * AD,
                      /*tile_d0=*/AD, /*tile_d1=*/KT, /*stride0=*/AD,
                      /*pad: every 32 dwords*/ 4, /*+4 dwords*/ 3);
      __builtin_amdgcn_s_wait_tensorcnt(0);
    }
#else
    {
      const v8h* src = (const v8h*)(Kp + (size_t)kt * KT * AD);
      v8h t[4];
#pragma unroll
      for (int it = 0; it < 4; ++it) t[it] = src[threadIdx.x + it * 128];
#pragma unroll
      for (int it = 0; it < 4; ++it) {
        const int i = threadIdx.x + it * 128;
        *(v8h*)&Kt[i >> 3][(i & 7) * 8] = t[it];
      }
    }
#endif
    __syncthreads();
#pragma unroll
    for (int nt = 0; nt < 4; ++nt) {
      const int krow = nt * 16 + l16;   // B col = local key
      const int cb = half * 8;
      v16h b0, b1;
#pragma unroll
      for (int e = 0; e < 16; ++e) {
        b0[e] = Kt[krow][cb + (e & 7) + (e >> 3) * 16];        // kk = table order (K=0..31 step)
        b1[e] = Kt[krow][32 + cb + (e & 7) + (e >> 3) * 16];   // kk = 32 + table order (K=32..63 step)
      }
      v8f s = {};
      s = wmma16(aq0, b0, s);
      s = wmma16(aq1, b1, s);
      const int key = kt * KT + nt * 16 + l16;
#pragma unroll
      for (int r = 0; r < 8; ++r)
        Lg[wave][half * 8 + r][key] = (_Float16)s[r];
    }
  }
  __syncthreads();

  // ---- Phase 2: per-row max & sum(exp); 2 lanes per row scan 1024 each
  {
    const int row = l16;
    const v8h* lv = (const v8h*)&Lg[wave][row][half * (S / 2)];
    float m = -3.0e38f;
    for (int i = 0; i < (S / 2) / 8; ++i) {
      const v8h c = lv[i];
#pragma unroll
      for (int j = 0; j < 8; ++j) m = fmaxf(m, (float)c[j]);
    }
    m = fmaxf(m, __shfl_xor(m, 16, 32));
    float sum = 0.f;
    for (int i = 0; i < (S / 2) / 8; ++i) {
      const v8h c = lv[i];
#pragma unroll
      for (int j = 0; j < 8; ++j) sum += __expf((float)c[j] - m);
    }
    sum += __shfl_xor(sum, 16, 32);
    if (half == 0) {
      RowMax[wave][row] = m;
      RowSum[wave][row] = sum;
    }
  }
  __syncthreads();

  // ---- Phase 3: write normalized attn once; ctx += P @ V (unnormalized P)
  const float am   = RowMax[wave][l16];       // stats for A-frag row = l16
  const float ainv = 1.0f / RowSum[wave][l16];
  v8f cacc[4] = {};
  const int koff = half * 8;
  float* arow_base = Attn + ((size_t)bh * S + (q0 + wave * 16 + l16)) * S;

  for (int kt = 0; kt < S / KT; ++kt) {
    __syncthreads();
#if USE_TDM
    if (wave == 0) {
      tdm_load_2d_f16((unsigned)(size_t)&Vt[0][0], Vp + (size_t)kt * KT,
                      /*tile_d0=*/KT, /*tile_d1=*/AD, /*stride0=*/S,
                      /*pad: every 32 dwords*/ 4, /*+4 dwords*/ 3);
      __builtin_amdgcn_s_wait_tensorcnt(0);
    }
#else
    {
      v8h t[4];
#pragma unroll
      for (int it = 0; it < 4; ++it) {
        const int i = threadIdx.x + it * 128;
        t[it] = *(const v8h*)&Vp[(size_t)(i >> 3) * S + kt * KT + (i & 7) * 8];
      }
#pragma unroll
      for (int it = 0; it < 4; ++it) {
        const int i = threadIdx.x + it * 128;
        *(v8h*)&Vt[i >> 3][(i & 7) * 8] = t[it];
      }
    }
#endif
    __syncthreads();

    v16h ap0, ap1;
    float* arow = arow_base + kt * KT;
    const _Float16* lrow = &Lg[wave][l16][kt * KT];
#pragma unroll
    for (int e = 0; e < 8; ++e) {
      const float p0 = __expf((float)lrow[koff + e] - am);
      const float p1 = __expf((float)lrow[16 + koff + e] - am);
      const float p2 = __expf((float)lrow[32 + koff + e] - am);
      const float p3 = __expf((float)lrow[48 + koff + e] - am);
      ap0[e]     = (_Float16)p0;
      ap0[e + 8] = (_Float16)p1;
      ap1[e]     = (_Float16)p2;
      ap1[e + 8] = (_Float16)p3;
      arow[koff + e]      = p0 * ainv;
      arow[16 + koff + e] = p1 * ainv;
      arow[32 + koff + e] = p2 * ainv;
      arow[48 + koff + e] = p3 * ainv;
    }
#pragma unroll
    for (int nt = 0; nt < 4; ++nt) {
      const int drow = nt * 16 + l16;   // B col = output dim
      const int cb = half * 8;
      v16h b0, b1;
#pragma unroll
      for (int e = 0; e < 16; ++e) {
        b0[e] = Vt[drow][cb + (e & 7) + (e >> 3) * 16];        // kk (key) in table order
        b1[e] = Vt[drow][32 + cb + (e & 7) + (e >> 3) * 16];
      }
      cacc[nt] = wmma16(ap0, b0, cacc[nt]);
      cacc[nt] = wmma16(ap1, b1, cacc[nt]);
    }
  }

  // ---- Epilogue: ctx / rowsum -> merged (b, s, h*AD) f16
  const int b = bh / H;
  const int h = bh % H;
#pragma unroll
  for (int r = 0; r < 8; ++r) {
    const int row16 = half * 8 + r;
    const float inv = 1.0f / RowSum[wave][row16];
    const int srow = q0 + wave * 16 + row16;
#pragma unroll
    for (int nt = 0; nt < 4; ++nt) {
      const int col = h * AD + nt * 16 + l16;
      Ctx[((size_t)(b * S + srow)) * D + col] = (_Float16)(cacc[nt][r] * inv);
    }
  }
}

// ---------------------------------------------------------------------------
// Output projection: x = ctx(f16) @ W_O(f32->f16), f32 out. ctx tile via TDM.
// ---------------------------------------------------------------------------
__global__ __launch_bounds__(128) void out_proj_kernel(
    const _Float16* __restrict__ Ctx, const float* __restrict__ W,
    float* __restrict__ Xout) {
  using namespace cfg;
  __shared__ _Float16 As[64][40];    // 64x32 ctx tile (+16B/row pad)
  __shared__ _Float16 BsT[64][40];   // 32x64 W tile, TRANSPOSED (n, k)
  const int m0 = blockIdx.y * 64;
  const int n0 = blockIdx.x * 64;
  const int lane = threadIdx.x & 31;
  const int wave = threadIdx.x >> 5;
  const int half = lane >> 4;
  const int l16 = lane & 15;

  v8f acc[4] = {};

  for (int k0 = 0; k0 < D; k0 += 32) {
    __syncthreads();
#if USE_TDM
    if (wave == 0) {
      tdm_load_2d_f16((unsigned)(size_t)&As[0][0], Ctx + (size_t)m0 * D + k0,
                      /*tile_d0=*/32, /*tile_d1=*/64, /*stride0=*/D,
                      /*pad: every 16 dwords*/ 3, /*+4 dwords*/ 3);
    }
#else
    {
      v8h t[2];
#pragma unroll
      for (int it = 0; it < 2; ++it) {
        const int i = threadIdx.x + it * 128;
        t[it] = *(const v8h*)&Ctx[(size_t)(m0 + (i >> 2)) * D + k0 +
                                  (i & 3) * 8];
      }
#pragma unroll
      for (int it = 0; it < 2; ++it) {
        const int i = threadIdx.x + it * 128;
        *(v8h*)&As[i >> 2][(i & 3) * 8] = t[it];
      }
    }
#endif
    {
      v4f wa[4];
#pragma unroll
      for (int it = 0; it < 4; ++it) {
        const int i = threadIdx.x + it * 128;
        wa[it] =
            *(const v4f*)&W[(size_t)(k0 + (i >> 4)) * D + n0 + (i & 15) * 4];
      }
#pragma unroll
      for (int it = 0; it < 4; ++it) {
        const int i = threadIdx.x + it * 128;
        const int br = i >> 4, bc = (i & 15) * 4;
        BsT[bc + 0][br] = (_Float16)wa[it][0];
        BsT[bc + 1][br] = (_Float16)wa[it][1];
        BsT[bc + 2][br] = (_Float16)wa[it][2];
        BsT[bc + 3][br] = (_Float16)wa[it][3];
      }
    }
#if USE_TDM
    if (wave == 0) __builtin_amdgcn_s_wait_tensorcnt(0);
#endif
    __syncthreads();

    const int arow = wave * 16 + l16;
    const int koff = half * 8;
    v16h af;
#pragma unroll
    for (int e = 0; e < 8; ++e) {
      af[e]     = As[arow][koff + e];
      af[e + 8] = As[arow][16 + koff + e];
    }
#pragma unroll
    for (int nt = 0; nt < 4; ++nt) {
      const int bcol = nt * 16 + l16;
      const int kb = half * 8;
      v16h bf;
#pragma unroll
      for (int e = 0; e < 16; ++e) bf[e] = BsT[bcol][kb + (e & 7) + (e >> 3) * 16];  // table order
      acc[nt] = wmma16(af, bf, acc[nt]);
    }
  }

#pragma unroll
  for (int nt = 0; nt < 4; ++nt) {
#pragma unroll
    for (int r = 0; r < 8; ++r) {
      const int m = m0 + wave * 16 + half * 8 + r;
      const int n = n0 + nt * 16 + l16;
      Xout[(size_t)m * D + n] = acc[nt][r];
    }
  }
}

// ---------------------------------------------------------------------------
extern "C" void kernel_launch(void* const* d_in, const int* in_sizes, int n_in,
                              void* d_out, int out_size, void* d_ws,
                              size_t ws_size, hipStream_t stream) {
  using namespace cfg;
  const float* q  = (const float*)d_in[0];
  const float* k  = (const float*)d_in[1];
  const float* v  = (const float*)d_in[2];
  const float* wq = (const float*)d_in[3];
  const float* wk = (const float*)d_in[4];
  const float* wv = (const float*)d_in[5];
  const float* wo = (const float*)d_in[6];

  float* x_out    = (float*)d_out;                 // (B, S, D) fp32
  float* attn_out = x_out + (size_t)BS * D;        // (B, H, S, S) fp32

  // f16 workspace: qh | kh (B,H,S,AD) | vhT (B,H,AD,S) | ctx (B,S,D) = 32 MB
  _Float16* ws  = (_Float16*)d_ws;
  _Float16* qh  = ws;
  _Float16* kh  = qh + (size_t)BS * D;
  _Float16* vhT = kh + (size_t)BS * D;
  _Float16* ctx = vhT + (size_t)BS * D;

  const dim3 pb(128), pg(D / 64, BS / 64);
  // Q/K: (S, AD) layout; scale 1/sqrt(AD) folded into Q.
  hipLaunchKernelGGL(qkv_proj_kernel, pg, pb, 0, stream, q, wq, qh, 0.125f,
                     /*dStride=*/1, /*sStride=*/AD);
  hipLaunchKernelGGL(qkv_proj_kernel, pg, pb, 0, stream, k, wk, kh, 1.0f,
                     /*dStride=*/1, /*sStride=*/AD);
  // V: transposed (AD, S) layout so attention V tiles are TDM-friendly.
  hipLaunchKernelGGL(qkv_proj_kernel, pg, pb, 0, stream, v, wv, vhT, 1.0f,
                     /*dStride=*/S, /*sStride=*/1);

  const dim3 ab(NW * 32), ag(S / (NW * 16), B * H);
  hipLaunchKernelGGL(attention_kernel, ag, ab, 0, stream, qh, kh, vhT,
                     attn_out, ctx);

  hipLaunchKernelGGL(out_proj_kernel, pg, pb, 0, stream, ctx, wo, x_out);

  (void)in_sizes; (void)n_in; (void)out_size; (void)ws_size;
}
